// TextLSTM_206158430775
// MI455X (gfx1250) — matrix operand order repaired and checked by reading, not compiled, not run on hardware
//
#include <hip/hip_runtime.h>
#include <hip/hip_bf16.h>
#include <stdint.h>

// ---------------------------------------------------------------------------
// Types for CDNA5 WMMA (wave32): D(16x16 f32) = A(16x32 bf16) x B(32x16 bf16) + C
// ---------------------------------------------------------------------------
typedef __attribute__((ext_vector_type(16))) __bf16 v16bf;
typedef __attribute__((ext_vector_type(8)))  __bf16 v8bf;
typedef __attribute__((ext_vector_type(8)))  float  v8f;

#define BATCH   64
#define SEQL    512
#define EMBED   512
#define HIDDEN  1024
#define K1      1536      // EMBED + HIDDEN
#define K2      2048      // HIDDEN + HIDDEN
#define NWG     64        // persistent workgroups (one 16-wide j slice each)
#define TPB     256       // 8 waves per WG

// ---------------- workspace layout (bytes, all 256-aligned) ----------------
static constexpr size_t SZ_EMB   = 32000ull * 512 * 2;        // emb bf16
static constexpr size_t SZ_W1    = 4ull * 1024 * 1536 * 2;    // W1 bf16 [n][k]
static constexpr size_t SZ_W2    = 4ull * 1024 * 2048 * 2;    // W2 bf16 [n][k]
static constexpr size_t SZ_H1BF  = 2ull * BATCH * HIDDEN * 2; // double buffered
static constexpr size_t SZ_H2BF  = 2ull * BATCH * HIDDEN * 2; // double buffered
static constexpr size_t SZ_C     = (size_t)BATCH * HIDDEN * 4;
static constexpr size_t SZ_SYNC  = 256;

static constexpr size_t OFF_EMB  = 0;
static constexpr size_t OFF_W1   = OFF_EMB + SZ_EMB;
static constexpr size_t OFF_W2   = OFF_W1 + SZ_W1;
static constexpr size_t OFF_H1BF = OFF_W2 + SZ_W2;            // zero region start
static constexpr size_t OFF_H2BF = OFF_H1BF + SZ_H1BF;
static constexpr size_t OFF_C1   = OFF_H2BF + SZ_H2BF;
static constexpr size_t OFF_C2   = OFF_C1 + SZ_C;
static constexpr size_t OFF_SYNC = OFF_C2 + SZ_C;
static constexpr size_t OFF_H2F  = OFF_SYNC + SZ_SYNC;        // fp32 h2 for FC
static constexpr size_t ZERO_BYTES = SZ_H1BF + SZ_H2BF + 2 * SZ_C + SZ_SYNC;

// ---------------------------------------------------------------------------
__device__ __forceinline__ v8f wmma_bf16(v16bf a, v16bf b, v8f c) {
  return __builtin_amdgcn_wmma_f32_16x16x32_bf16(
      /*neg_a=*/false, a, /*neg_b=*/false, b,
      /*c_mod=*/(short)0, c, /*reuse_a=*/false, /*reuse_b=*/false);
}

// A fragment (16-bit A 16x32): lane<16 holds K {0..7,16..23}, lane>=16 holds
// K {8..15,24..31}. p already includes the lane-half offset (lh*8 elements).
__device__ __forceinline__ v16bf make_afrag(const __bf16* p) {
  union { v16bf v; v8bf h[2]; } u;
  u.h[0] = *reinterpret_cast<const v8bf*>(p);        // K + 0..7
  u.h[1] = *reinterpret_cast<const v8bf*>(p + 16);   // K + 16..23
  return u.v;
}

// B fragment in contiguous k order (16h+s; unused, gemm_region loads B via make_afrag): lanes 0-15 hold K=0..15 of column N=lane,
// lanes 16-31 hold K=16..31 -> one contiguous 32-byte load per lane.
__device__ __forceinline__ v16bf make_bfrag(const __bf16* p) {
  return *reinterpret_cast<const v16bf*>(p);
}

__device__ __forceinline__ float sigm(float x) {
  return 1.0f / (1.0f + __expf(-x));
}

// Branch-free GEMM slice over one contiguous K region: both A streams and the
// B (weight) stream advance linearly; 2 WMMAs per 32-wide K chunk.
__device__ __forceinline__ void gemm_region(const __bf16* __restrict__ a0,
                                            const __bf16* __restrict__ a1,
                                            const __bf16* __restrict__ brow,
                                            int klen, int lh,
                                            v8f& acc0, v8f& acc1) {
#pragma unroll 4
  for (int k0 = 0; k0 < klen; k0 += 32) {
    __builtin_prefetch(brow + k0 + 256, 0, 1);            // next weight chunks
    v16bf bf = make_afrag(brow + k0 + lh * 8);
    v16bf af0 = make_afrag(a0 + k0);
    v16bf af1 = make_afrag(a1 + k0);
    acc0 = wmma_bf16(af0, bf, acc0);
    acc1 = wmma_bf16(af1, bf, acc1);
  }
}

// Grid-wide barrier via global atomics (persistent-kernel sync).
__device__ __forceinline__ void grid_barrier(int* cnt, int* gen) {
  __syncthreads();
  if (threadIdx.x == 0) {
    __threadfence();
    int g = atomicAdd(gen, 0);
    if (atomicAdd(cnt, 1) == NWG - 1) {
      atomicExch(cnt, 0);
      __threadfence();
      atomicAdd(gen, 1);
    } else {
      while (atomicAdd(gen, 0) == g) { __builtin_amdgcn_s_sleep(1); }
    }
    __threadfence();
  }
  __syncthreads();
}

// ---------------------------------------------------------------------------
__global__ void cvt_bf16_kernel(const float* __restrict__ src,
                                __bf16* __restrict__ dst, int n) {
  int i = blockIdx.x * blockDim.x + threadIdx.x;
  int stride = gridDim.x * blockDim.x;
  for (; i < n; i += stride) dst[i] = (__bf16)src[i];
}

__global__ void zero_kernel(uint32_t* __restrict__ p, int nwords) {
  int i = blockIdx.x * blockDim.x + threadIdx.x;
  int stride = gridDim.x * blockDim.x;
  for (; i < nwords; i += stride) p[i] = 0u;
}

// ---------------------------------------------------------------------------
// Persistent 2-layer LSTM. WG b owns gate columns j in [16b, 16b+16) for all
// 4 gates and all 4 M-tiles. Wave v: gate = v>>1, M-tiles {2(v&1), 2(v&1)+1}.
// ---------------------------------------------------------------------------
__global__ __launch_bounds__(TPB, 1)
void lstm_persistent_kernel(const int*    __restrict__ x,
                            const __bf16* __restrict__ emb,
                            const __bf16* __restrict__ w1,
                            const __bf16* __restrict__ w2,
                            const float*  __restrict__ b1,
                            const float*  __restrict__ b2,
                            __bf16* h1bf, __bf16* h2bf,
                            float* c1, float* c2,
                            float* h2f, int* sync) {
  __shared__ float gtile[64 * 64];   // 64 batch rows x (4 gates x 16 j) fp32

  const int wave = threadIdx.x >> 5;
  const int lane = threadIdx.x & 31;
  const int lh   = lane >> 4;        // lane half
  const int l15  = lane & 15;
  const int koff = lh * 8;           // A-fragment K offset for this half
  const int j0   = blockIdx.x * 16;
  const int gate = wave >> 1;
  const int mt0  = (wave & 1) * 2;   // first of two M-tiles
  const int rowA0 = mt0 * 16 + l15;  // batch rows for A fragments
  const int rowA1 = rowA0 + 16;
  const int nrow  = gate * HIDDEN + j0 + l15;      // weight row (gate col)
  const __bf16* w1row = w1 + (size_t)nrow * K1;
  const __bf16* w2row = w2 + (size_t)nrow * K2;
  // recurrent-state row bases (plus per-lane-half A offset), per buffer
  const __bf16* h1r0[2] = { h1bf + rowA0 * HIDDEN + koff,
                            h1bf + (size_t)BATCH * HIDDEN + rowA0 * HIDDEN + koff };
  const __bf16* h1r1[2] = { h1bf + rowA1 * HIDDEN + koff,
                            h1bf + (size_t)BATCH * HIDDEN + rowA1 * HIDDEN + koff };
  const __bf16* h2r0[2] = { h2bf + rowA0 * HIDDEN + koff,
                            h2bf + (size_t)BATCH * HIDDEN + rowA0 * HIDDEN + koff };
  const __bf16* h2r1[2] = { h2bf + rowA1 * HIDDEN + koff,
                            h2bf + (size_t)BATCH * HIDDEN + rowA1 * HIDDEN + koff };
  const int colc = gate * 16 + l15;                // LDS column
  int* cnt = sync;
  int* gen = sync + 1;

  for (int t = 0; t < SEQL; ++t) {
    const int rb = t & 1;            // read buffer for h1/h2 state
    const int wb = rb ^ 1;           // write buffer

    // ---------------- layer 1: g1 = [emb[x_t] | h1_{t-1}] x W1^T ----------
    {
      const int tok0 = x[rowA0 * SEQL + t];
      const int tok1 = x[rowA1 * SEQL + t];
      v8f acc0 = {}, acc1 = {};
      // K region 0: embedding gather rows
      gemm_region(emb + (size_t)tok0 * EMBED + koff,
                  emb + (size_t)tok1 * EMBED + koff,
                  w1row, EMBED, lh, acc0, acc1);
      // K region 1: recurrent h1
      gemm_region(h1r0[rb], h1r1[rb], w1row + EMBED, HIDDEN, lh, acc0, acc1);
#pragma unroll
      for (int r = 0; r < 8; ++r) {
        gtile[(mt0 * 16 + lh * 8 + r) * 64 + colc]       = acc0[r];
        gtile[((mt0 + 1) * 16 + lh * 8 + r) * 64 + colc] = acc1[r];
      }
    }
    __syncthreads();

    // gate nonlinearities, layer 1 (WG-local: all 4 gates live in gtile)
    for (int it = threadIdx.x; it < 1024; it += TPB) {
      const int b = it >> 4, jj = it & 15, j = j0 + jj;
      const float gf = gtile[b * 64 +  0 + jj] + b1[0 * HIDDEN + j];
      const float gi = gtile[b * 64 + 16 + jj] + b1[1 * HIDDEN + j];
      const float gc = gtile[b * 64 + 32 + jj] + b1[2 * HIDDEN + j];
      const float go = gtile[b * 64 + 48 + jj] + b1[3 * HIDDEN + j];
      const float cn = sigm(gf) * c1[b * HIDDEN + j] + sigm(gi) * tanhf(gc);
      c1[b * HIDDEN + j] = cn;
      const float h = sigm(go) * tanhf(cn);
      h1bf[(size_t)wb * BATCH * HIDDEN + b * HIDDEN + j] = (__bf16)h;
    }
    grid_barrier(cnt, gen);   // h1_t visible to every WG

    // ---------------- layer 2: g2 = [h1_t | h2_{t-1}] x W2^T --------------
    {
      v8f acc0 = {}, acc1 = {};
      // K region 0: h1_t (freshly written buffer wb)
      gemm_region(h1r0[wb], h1r1[wb], w2row, HIDDEN, lh, acc0, acc1);
      // K region 1: recurrent h2
      gemm_region(h2r0[rb], h2r1[rb], w2row + HIDDEN, HIDDEN, lh, acc0, acc1);
#pragma unroll
      for (int r = 0; r < 8; ++r) {
        gtile[(mt0 * 16 + lh * 8 + r) * 64 + colc]       = acc0[r];
        gtile[((mt0 + 1) * 16 + lh * 8 + r) * 64 + colc] = acc1[r];
      }
    }
    __syncthreads();

    // gate nonlinearities, layer 2
    for (int it = threadIdx.x; it < 1024; it += TPB) {
      const int b = it >> 4, jj = it & 15, j = j0 + jj;
      const float gf = gtile[b * 64 +  0 + jj] + b2[0 * HIDDEN + j];
      const float gi = gtile[b * 64 + 16 + jj] + b2[1 * HIDDEN + j];
      const float gc = gtile[b * 64 + 32 + jj] + b2[2 * HIDDEN + j];
      const float go = gtile[b * 64 + 48 + jj] + b2[3 * HIDDEN + j];
      const float cn = sigm(gf) * c2[b * HIDDEN + j] + sigm(gi) * tanhf(gc);
      c2[b * HIDDEN + j] = cn;
      const float h = sigm(go) * tanhf(cn);
      h2f[b * HIDDEN + j] = h;
      h2bf[(size_t)wb * BATCH * HIDDEN + b * HIDDEN + j] = (__bf16)h;
    }
    grid_barrier(cnt, gen);   // h2_t (and next-step state) visible everywhere
  }
}

// ---------------------------------------------------------------------------
__global__ void fc_kernel(const float* __restrict__ h2,
                          const float* __restrict__ wfc,
                          const float* __restrict__ bfc,
                          float* __restrict__ out) {
  const int idx = blockIdx.x * blockDim.x + threadIdx.x;
  if (idx >= BATCH * 10) return;
  const int b = idx / 10, cls = idx % 10;
  const float* hr = h2 + b * HIDDEN;
  const float* wr = wfc + cls * HIDDEN;
  float s = bfc[cls];
  for (int k = 0; k < HIDDEN; ++k) s += hr[k] * wr[k];
  out[idx] = s;
}

// ---------------------------------------------------------------------------
extern "C" void kernel_launch(void* const* d_in, const int* in_sizes, int n_in,
                              void* d_out, int out_size, void* d_ws, size_t ws_size,
                              hipStream_t stream) {
  (void)in_sizes; (void)n_in; (void)out_size; (void)ws_size;
  const int*   x   = (const int*)d_in[0];
  const float* emb = (const float*)d_in[1];
  const float* W1  = (const float*)d_in[2];
  const float* b1  = (const float*)d_in[3];
  const float* W2  = (const float*)d_in[4];
  const float* b2  = (const float*)d_in[5];
  const float* Wfc = (const float*)d_in[6];
  const float* bfc = (const float*)d_in[7];

  char* ws = (char*)d_ws;
  __bf16* embbf = (__bf16*)(ws + OFF_EMB);
  __bf16* w1bf  = (__bf16*)(ws + OFF_W1);
  __bf16* w2bf  = (__bf16*)(ws + OFF_W2);
  __bf16* h1bf  = (__bf16*)(ws + OFF_H1BF);
  __bf16* h2bf  = (__bf16*)(ws + OFF_H2BF);
  float*  c1    = (float*)(ws + OFF_C1);
  float*  c2    = (float*)(ws + OFF_C2);
  int*    sync  = (int*)(ws + OFF_SYNC);
  float*  h2f   = (float*)(ws + OFF_H2F);

  // one-time prep (re-run every call: deterministic, ws may be poisoned)
  cvt_bf16_kernel<<<2048, 256, 0, stream>>>(emb, embbf, 32000 * 512);
  cvt_bf16_kernel<<<1024, 256, 0, stream>>>(W1,  w1bf, 4 * 1024 * 1536);
  cvt_bf16_kernel<<<1024, 256, 0, stream>>>(W2,  w2bf, 4 * 1024 * 2048);
  zero_kernel<<<256, 256, 0, stream>>>((uint32_t*)(ws + OFF_H1BF),
                                       (int)(ZERO_BYTES / 4));

  lstm_persistent_kernel<<<NWG, TPB, 0, stream>>>(x, embbf, w1bf, w2bf,
                                                  b1, b2, h1bf, h2bf,
                                                  c1, c2, h2f, sync);

  fc_kernel<<<3, 256, 0, stream>>>(h2f, Wfc, bfc, (float*)d_out);
}
